// GATCrossAttentionPretrainPI_81235011437205
// MI455X (gfx1250) — compile-verified, matrix operand order checked by reading, not run on hardware
//
#include <hip/hip_runtime.h>
#include <hip/hip_bf16.h>

// ---------------------------------------------------------------------------
// GAT + cross-attention GNN pipeline for MI455X (gfx1250, wave32, WMMA).
// GEMMs run on v_wmma_f32_16x16x32_f16 (f16 in, fp32 accumulate) with
// fragment-major pre-swizzled weights so every WMMA operand is fed by
// wide b128 loads (2x ds_load_b128 or 2x global_load_b128 per fragment).
// Edge softmax/aggregation (the bandwidth bound: ~13 GB of gather/scatter
// at 23.3 TB/s) runs on float4 per-lane loads + f32 atomics.
// ---------------------------------------------------------------------------

#define N_NODES 100000
#define N_EDGES 1600000
#define BGR 512

typedef __attribute__((ext_vector_type(16))) _Float16 v16h;
typedef __attribute__((ext_vector_type(8)))  float    v8f;

union Frag { v16h v; uint4 q[2]; _Float16 e[16]; };

// K offset inside a 32-wide chunk for fragment dword v (0..7), per the
// documented 16-bit A/B 16x32 striping (lanes 0-15 / 16-31 = K halves).
__device__ __forceinline__ int frag_k(int v, int half) {
  return ((v & 4) << 2) + (half << 3) + ((v & 3) << 1);
}

// Monotone float->uint encoding so atomicMax(uint) implements float max.
__device__ __forceinline__ unsigned fenc(float x) {
  unsigned u = __float_as_uint(x);
  return (u & 0x80000000u) ? ~u : (u | 0x80000000u);
}
__device__ __forceinline__ float fdec(unsigned e) {
  unsigned u = (e & 0x80000000u) ? (e & 0x7FFFFFFFu) : ~e;
  return __uint_as_float(u);
}

// ---------------------------------------------------------------------------
// Weight fp32 -> f16 conversion into FRAGMENT-MAJOR layout:
// dst[((chunk*ntiles16 + tile)*32 + lane)*16 + e] with e indexing the lane's
// packed 16 halfs of the B operand for (k-chunk, 16-col tile). This lets the
// GEMMs fetch each B fragment as two contiguous 16-byte loads per lane.
// ---------------------------------------------------------------------------
__global__ void convert_w_frag_kernel(const float* __restrict__ w, _Float16* __restrict__ dst,
                                      int K, int Nc, int Kpad, int Npad)
{
  long long i = (long long)blockIdx.x * 256 + threadIdx.x;
  if (i >= (long long)Kpad * Npad) return;
  int e    = (int)(i & 15);
  int lane = (int)((i >> 4) & 31);
  long long fidx = i >> 9;                // chunk*ntiles16 + tile
  int ntiles16 = Npad >> 4;
  int tile  = (int)(fidx % ntiles16);
  int chunk = (int)(fidx / ntiles16);
  int half = lane >> 4;
  int c = tile * 16 + (lane & 15);
  int k = chunk * 32 + frag_k(e >> 1, half) + (e & 1);
  float val = (k < K && c < Nc) ? w[(size_t)k * Nc + c] : 0.f;
  dst[i] = (_Float16)val;
}

// fp32 -> f16 row-major with zero padding (activations; A fragments are
// contiguous 16B runs in row-major layout so no swizzle needed)
__global__ void convert_pad_kernel(const float* __restrict__ src, _Float16* __restrict__ dst,
                                   int rows, int cols, int rowsPad, int colsPad)
{
  long long i = (long long)blockIdx.x * 256 + threadIdx.x;
  if (i >= (long long)rowsPad * colsPad) return;
  int r = (int)(i / colsPad), c = (int)(i % colsPad);
  float v = (r < rows && c < cols) ? src[(size_t)r * cols + c] : 0.f;
  dst[i] = (_Float16)v;
}

__global__ void fill_u32_kernel(unsigned* __restrict__ p, unsigned val, long long count)
{
  long long i = (long long)blockIdx.x * 256 + threadIdx.x;
  if (i < count) p[i] = val;
}

// ---------------------------------------------------------------------------
// GAT GEMM: hW = A(f16, n x K) @ W(fragment-packed f16, K x 128).
// Packed W staged into LDS with a linear uint4 copy; each wave computes a
// 16x128 tile (8 WMMA tiles). B fragments: 2x ds_load_b128 per lane.
// Epilogue fuses s_src = hW.a_src, s_dst = hW.a_dst via shfl_xor.
// ---------------------------------------------------------------------------
__global__ void __launch_bounds__(256)
gat_gemm_kernel(const _Float16* __restrict__ A, const _Float16* __restrict__ W,
                const float* __restrict__ a_src, const float* __restrict__ a_dst,
                float* __restrict__ hW, float* __restrict__ sS, float* __restrict__ sD,
                int n, int npad, int K)
{
  __shared__ _Float16 Wl[160 * 128];
  int tid = threadIdx.x;
  {
    int nq = K * 16;  // K*128 halfs / 8 per uint4
    uint4* d = (uint4*)Wl;
    const uint4* s = (const uint4*)W;
    for (int i = tid; i < nq; i += 256) d[i] = s[i];
  }
  __syncthreads();

  int wave = tid >> 5;
  int lane = tid & 31;
  int half = lane >> 4;
  int lc   = lane & 15;
  int row0 = (blockIdx.x * 8 + wave) * 16;
  if (row0 >= npad) return;

  v8f acc[8] = {};

  const _Float16* arow = A + (size_t)(row0 + lc) * K;
  for (int k0 = 0; k0 < K; k0 += 32) {
    Frag a;
    const uint4* ap = (const uint4*)(arow + k0 + half * 8);
    a.q[0] = ap[0];      // K = k0 + half*8 .. +7
    a.q[1] = ap[2];      // K = k0 + 16 + half*8 .. +7
#pragma unroll
    for (int t = 0; t < 8; ++t) {
      Frag b;
      const uint4* bp = (const uint4*)(Wl + ((((k0 >> 5) * 8 + t) * 32 + lane) << 4));
      b.q[0] = bp[0];
      b.q[1] = bp[1];
      acc[t] = __builtin_amdgcn_wmma_f32_16x16x32_f16(
          false, a.v, false, b.v, (short)0, acc[t], false, false);
    }
  }

#pragma unroll
  for (int j = 0; j < 8; ++j) {
    int r = row0 + j + 8 * half;
    float ps = 0.f, pd = 0.f;
#pragma unroll
    for (int t = 0; t < 8; ++t) {
      float c = acc[t][j];
      int cc = t * 16 + lc;
      if (r < n) hW[(size_t)r * 128 + cc] = c;
      ps += c * a_src[cc];
      pd += c * a_dst[cc];
    }
#pragma unroll
    for (int m = 1; m < 16; m <<= 1) {
      ps += __shfl_xor(ps, m, 32);
      pd += __shfl_xor(pd, m, 32);
    }
    if (lc == 0 && r < n) { sS[r] = ps; sD[r] = pd; }
  }
}

// ---------------------------------------------------------------------------
// Generic WMMA GEMM: out = act(A @ W + b), W fragment-packed in global.
// Wave = 16x64 tile; B fragments: 2x global_load_b128 per lane.
// act: 0=none 1=relu 2=elu.
// ---------------------------------------------------------------------------
__global__ void __launch_bounds__(256)
wmma_gemm_kernel(const _Float16* __restrict__ A, const _Float16* __restrict__ Wm,
                 const float* __restrict__ bias,
                 float* __restrict__ outF, int ldoF,
                 _Float16* __restrict__ outH, int ldoH,
                 int M, int Nc, int Mpad, int Npad, int K, int act)
{
  int tid = threadIdx.x;
  int wg = blockIdx.x * 8 + (tid >> 5);
  int lane = tid & 31;
  int half = lane >> 4;
  int lc = lane & 15;
  int ntiles64 = (Npad + 63) >> 6;
  int ntiles16 = Npad >> 4;
  int mt = wg / ntiles64;
  int ct = wg % ntiles64;
  int row0 = mt * 16;
  if (row0 >= Mpad) return;
  int tilebase = ct * 4;   // first 16-col tile index

  v8f acc[4] = {};

  const _Float16* arow = A + (size_t)(row0 + lc) * K;
  for (int k0 = 0; k0 < K; k0 += 32) {
    Frag a;
    const uint4* ap = (const uint4*)(arow + k0 + half * 8);
    a.q[0] = ap[0];
    a.q[1] = ap[2];
#pragma unroll
    for (int t = 0; t < 4; ++t) {
      int tile = tilebase + t;
      if (tile >= ntiles16) break;
      Frag b;
      const uint4* bp = (const uint4*)(Wm +
          ((((size_t)(k0 >> 5) * ntiles16 + tile) * 32 + lane) << 4));
      b.q[0] = bp[0];
      b.q[1] = bp[1];
      acc[t] = __builtin_amdgcn_wmma_f32_16x16x32_f16(
          false, a.v, false, b.v, (short)0, acc[t], false, false);
    }
  }

#pragma unroll
  for (int t = 0; t < 4; ++t) {
    int tile = tilebase + t;
    if (tile >= ntiles16) break;
    int c = tile * 16 + lc;
#pragma unroll
    for (int j = 0; j < 8; ++j) {
      int r = row0 + j + 8 * half;
      float v = acc[t][j];
      bool valid = (r < M) && (c < Nc);
      if (valid) {
        if (bias) v += bias[c];
        if (act == 1) v = v > 0.f ? v : 0.f;
        else if (act == 2) v = v > 0.f ? v : (__expf(v) - 1.f);
        if (outF) outF[(size_t)r * ldoF + c] = v;
      }
      if (outH) outH[(size_t)r * ldoH + c] = valid ? (_Float16)v : (_Float16)0.f;
    }
  }
}

// ---- Edge softmax / aggregation ------------------------------------------
__global__ void edge_score_kernel(const int* __restrict__ src, const int* __restrict__ dst,
                                  const float* __restrict__ sS, const float* __restrict__ sD,
                                  float* __restrict__ e, unsigned* __restrict__ mEnc, int E)
{
  int i = blockIdx.x * 256 + threadIdx.x;
  if (i >= E) return;
  int s = src[i], d = dst[i];
  float v = sS[s] + sD[d];
  v = v > 0.f ? v : 0.2f * v;          // leaky_relu 0.2
  e[i] = v;
  atomicMax(&mEnc[d], fenc(v));
}

__global__ void edge_exp_kernel(const int* __restrict__ dst, float* __restrict__ e,
                                const unsigned* __restrict__ mEnc, float* __restrict__ den, int E)
{
  int i = blockIdx.x * 256 + threadIdx.x;
  if (i >= E) return;
  int d = dst[i];
  float ex = __expf(e[i] - fdec(mEnc[d]));
  e[i] = ex;
  atomicAdd(&den[d], ex);
}

__global__ void __launch_bounds__(256)
edge_agg_kernel(const int* __restrict__ src, const int* __restrict__ dst,
                const float* __restrict__ ex, const float* __restrict__ den,
                const float* __restrict__ hW, float* __restrict__ out, int E)
{
  int wid = (blockIdx.x * 256 + threadIdx.x) >> 5;
  int lane = threadIdx.x & 31;
  if (wid >= E) return;
  int s = src[wid], d = dst[wid];
  float alpha = ex[wid] / (den[d] + 1e-16f);
  const float4 hv = *reinterpret_cast<const float4*>(hW + (size_t)s * 128 + lane * 4);
  float* o = out + (size_t)d * 128 + lane * 4;
  atomicAdd(o + 0, alpha * hv.x);
  atomicAdd(o + 1, alpha * hv.y);
  atomicAdd(o + 2, alpha * hv.z);
  atomicAdd(o + 3, alpha * hv.w);
}

__global__ void bias_elu_kernel(const float* __restrict__ agg, const float* __restrict__ bias,
                                float* __restrict__ hf, _Float16* __restrict__ h16,
                                int n, int npad)
{
  long long i = (long long)blockIdx.x * 256 + threadIdx.x;
  if (i >= (long long)npad * 128) return;
  int r = (int)(i >> 7), c = (int)(i & 127);
  float v = 0.f;
  if (r < n) {
    v = agg[i] + bias[c];
    v = v > 0.f ? v : (__expf(v) - 1.f);   // ELU
    hf[i] = v;
  }
  if (h16) h16[i] = (_Float16)v;
}

__global__ void concat_xf_kernel(const float* __restrict__ x, const float* __restrict__ f,
                                 _Float16* __restrict__ dst, int n, int npad)
{
  long long i = (long long)blockIdx.x * 256 + threadIdx.x;
  if (i >= (long long)npad * 160) return;
  int r = (int)(i / 160), c = (int)(i % 160);
  float v = 0.f;
  if (r < n) v = (c < 32) ? x[(size_t)r * 32 + c] : f[(size_t)r * 128 + (c - 32)];
  dst[i] = (_Float16)v;
}

// ---- Pooling --------------------------------------------------------------
__global__ void count_kernel(const int* __restrict__ batch, float* __restrict__ counts, int n)
{
  int i = blockIdx.x * 256 + threadIdx.x;
  if (i < n) atomicAdd(&counts[batch[i]], 1.0f);
}

__global__ void __launch_bounds__(256)
pool_accum_kernel(const int* __restrict__ batch, const float* __restrict__ h,
                  float* __restrict__ pool, int n)
{
  int wid = (blockIdx.x * 256 + threadIdx.x) >> 5;
  int lane = threadIdx.x & 31;
  if (wid >= n) return;
  int b = batch[wid];
  const float4 hv = *reinterpret_cast<const float4*>(h + (size_t)wid * 128 + lane * 4);
  float* p = pool + (size_t)b * 128 + lane * 4;
  atomicAdd(p + 0, hv.x); atomicAdd(p + 1, hv.y);
  atomicAdd(p + 2, hv.z); atomicAdd(p + 3, hv.w);
}

__global__ void pool_fin_kernel(float* __restrict__ pool, const float* __restrict__ counts,
                                _Float16* __restrict__ a16)
{
  int i = blockIdx.x * 256 + threadIdx.x;
  if (i >= BGR * 128) return;
  float v = pool[i] / fmaxf(counts[i >> 7], 1.0f);
  pool[i] = v;
  a16[i] = (_Float16)v;
}

// ---- Cross attention ------------------------------------------------------
__global__ void __launch_bounds__(256)
attn_score_kernel(const float* __restrict__ q, const float* __restrict__ k,
                  const int* __restrict__ batch, float* __restrict__ score,
                  unsigned* __restrict__ mEnc, int n)
{
  int wid = (blockIdx.x * 256 + threadIdx.x) >> 5;
  int lane = threadIdx.x & 31;
  if (wid >= n) return;
  int b = batch[wid];
  const float4 qv = *reinterpret_cast<const float4*>(q + (size_t)b * 128 + lane * 4);
  const float4 kv = *reinterpret_cast<const float4*>(k + (size_t)wid * 128 + lane * 4);
  float p = qv.x * kv.x + qv.y * kv.y + qv.z * kv.z + qv.w * kv.w;
#pragma unroll
  for (int m = 1; m < 32; m <<= 1) p += __shfl_xor(p, m, 32);
  if (lane == 0) {
    p *= 0.08838834764831845f;   // 1/sqrt(128)
    score[wid] = p;
    atomicMax(&mEnc[b], fenc(p));
  }
}

__global__ void attn_exp_kernel(const int* __restrict__ batch, float* __restrict__ score,
                                const unsigned* __restrict__ mEnc, float* __restrict__ den, int n)
{
  int i = blockIdx.x * 256 + threadIdx.x;
  if (i >= n) return;
  int b = batch[i];
  float ex = __expf(score[i] - fdec(mEnc[b]));
  score[i] = ex;
  atomicAdd(&den[b], ex);
}

__global__ void __launch_bounds__(256)
attn_agg_kernel(const int* __restrict__ batch, const float* __restrict__ ex,
                const float* __restrict__ den, const float* __restrict__ v,
                float* __restrict__ attn, int n)
{
  int wid = (blockIdx.x * 256 + threadIdx.x) >> 5;
  int lane = threadIdx.x & 31;
  if (wid >= n) return;
  int b = batch[wid];
  float alpha = ex[wid] / (den[b] + 1e-16f);
  const float4 vv = *reinterpret_cast<const float4*>(v + (size_t)wid * 128 + lane * 4);
  float* a = attn + (size_t)b * 128 + lane * 4;
  atomicAdd(a + 0, alpha * vv.x); atomicAdd(a + 1, alpha * vv.y);
  atomicAdd(a + 2, alpha * vv.z); atomicAdd(a + 3, alpha * vv.w);
}

__global__ void combine_kernel(const float* __restrict__ pi, const float* __restrict__ attn,
                               const float* __restrict__ me, _Float16* __restrict__ dst)
{
  int i = blockIdx.x * 256 + threadIdx.x;
  if (i >= BGR * 288) return;
  int r = i / 288, c = i % 288;
  float v = 0.f;
  if (c < 10)       v = pi[r * 10 + c];
  else if (c < 138) v = attn[r * 128 + (c - 10)];
  else if (c < 266) v = me[r * 128 + (c - 138)];
  dst[i] = (_Float16)v;
}

__global__ void final_kernel(const float* __restrict__ energy, const float* __restrict__ pi,
                             float* __restrict__ out)
{
  int i = blockIdx.x * 256 + threadIdx.x;
  if (i >= BGR + BGR * 10) return;
  out[i] = (i < BGR) ? energy[i] : pi[i - BGR];
}

// ---------------------------------------------------------------------------
// Host-side orchestration
// ---------------------------------------------------------------------------
static inline unsigned gdiv(long long a, long long b) { return (unsigned)((a + b - 1) / b); }

static void run_gemm(hipStream_t st, _Float16* Wbuf,
                     const _Float16* A, const float* Wsrc, const float* bias,
                     int Kact, int Kpad, int Nc, int Npad,
                     int M, int Mpad, int act,
                     float* outF, int ldoF, _Float16* outH, int ldoH)
{
  long long wtot = (long long)Kpad * Npad;
  convert_w_frag_kernel<<<gdiv(wtot, 256), 256, 0, st>>>(Wsrc, Wbuf, Kact, Nc, Kpad, Npad);
  int ntiles64 = (Npad + 63) / 64;
  long long waves = (long long)(Mpad / 16) * ntiles64;
  wmma_gemm_kernel<<<gdiv(waves, 8), 256, 0, st>>>(A, Wbuf, bias, outF, ldoF, outH, ldoH,
                                                   M, Nc, Mpad, Npad, Kpad, act);
}

static void run_gat_layer(hipStream_t st, const _Float16* Ain, int K,
                          const float* Wsrc, const float* a_src, const float* a_dst,
                          const float* bsrc, _Float16* Wbuf,
                          float* hW, float* sS, float* sD,
                          unsigned* mEnc, float* den, float* earr, float* agg,
                          const int* src, const int* dst,
                          float* hf, _Float16* h16out)
{
  convert_w_frag_kernel<<<gdiv((long long)K * 128, 256), 256, 0, st>>>(Wsrc, Wbuf, K, 128, K, 128);
  gat_gemm_kernel<<<gdiv(N_NODES, 128), 256, 0, st>>>(Ain, Wbuf, a_src, a_dst,
                                                      hW, sS, sD, N_NODES, N_NODES, K);
  fill_u32_kernel<<<gdiv(N_NODES, 256), 256, 0, st>>>(mEnc, 0u, N_NODES);
  fill_u32_kernel<<<gdiv(N_NODES, 256), 256, 0, st>>>((unsigned*)den, 0u, N_NODES);
  fill_u32_kernel<<<gdiv((long long)N_NODES * 128, 256), 256, 0, st>>>((unsigned*)agg, 0u,
                                                                       (long long)N_NODES * 128);
  edge_score_kernel<<<gdiv(N_EDGES, 256), 256, 0, st>>>(src, dst, sS, sD, earr, mEnc, N_EDGES);
  edge_exp_kernel<<<gdiv(N_EDGES, 256), 256, 0, st>>>(dst, earr, mEnc, den, N_EDGES);
  edge_agg_kernel<<<gdiv(N_EDGES, 8), 256, 0, st>>>(src, dst, earr, den, hW, agg, N_EDGES);
  bias_elu_kernel<<<gdiv((long long)N_NODES * 128, 256), 256, 0, st>>>(agg, bsrc, hf, h16out,
                                                                       N_NODES, N_NODES);
}

extern "C" void kernel_launch(void* const* d_in, const int* in_sizes, int n_in,
                              void* d_out, int out_size, void* d_ws, size_t ws_size,
                              hipStream_t stream)
{
  const float* x      = (const float*)d_in[0];
  const int*   ei     = (const int*)d_in[1];
  const int*   src    = ei;
  const int*   dstp   = ei + N_EDGES;
  const int*   batch  = (const int*)d_in[2];
  const float* metal  = (const float*)d_in[3];
  // params flattened in setup_inputs() insertion order:
  // node_layers[4]{W,a_src,a_dst,b}, gat_layers[4]{...}, metal_fc{w1,b1,w2,b2},
  // Wq{w,b}, Wk{w,b}, Wv{w,b}, pi_fc[5]{w,b}, regressor[5]{w,b}
  auto P = [&](int i) { return (const float*)d_in[i]; };

  // --- workspace carve (deterministic) ---
  char* w = (char*)d_ws;
  size_t off = 0;
  auto carve = [&](size_t bytes) {
    void* p = w + off;
    off = (off + bytes + 255) & ~(size_t)255;
    return p;
  };
  _Float16* actA  = (_Float16*)carve((size_t)N_NODES * 160 * 2);
  _Float16* Wbuf  = (_Float16*)carve((size_t)512 * 512 * 2);
  float* hW       = (float*)carve((size_t)N_NODES * 128 * 4);   // later: k
  float* agg      = (float*)carve((size_t)N_NODES * 128 * 4);   // later: v
  float* hf       = (float*)carve((size_t)N_NODES * 128 * 4);
  float* sS       = (float*)carve((size_t)N_NODES * 4);         // later: attn scores
  float* sD       = (float*)carve((size_t)N_NODES * 4);
  unsigned* mEnc  = (unsigned*)carve((size_t)N_NODES * 4);
  float* den      = (float*)carve((size_t)N_NODES * 4);
  float* earr     = (float*)carve((size_t)N_EDGES * 4);
  float* pool     = (float*)carve((size_t)BGR * 128 * 4);
  float* counts   = (float*)carve((size_t)BGR * 4);
  _Float16* bufB1 = (_Float16*)carve((size_t)512 * 512 * 2);
  _Float16* bufB2 = (_Float16*)carve((size_t)512 * 512 * 2);
  float* pi_out   = (float*)carve((size_t)BGR * 10 * 4);
  float* membed   = (float*)carve((size_t)BGR * 128 * 4);
  float* qbuf     = (float*)carve((size_t)BGR * 128 * 4);
  float* attnB    = (float*)carve((size_t)BGR * 128 * 4);
  unsigned* mB    = (unsigned*)carve((size_t)BGR * 4);
  float* denB     = (float*)carve((size_t)BGR * 4);
  float* energy   = (float*)carve((size_t)BGR * 4);
  (void)ws_size; (void)in_sizes; (void)n_in; (void)out_size;

  // --- node_layers chain (x -> f) ---
  convert_pad_kernel<<<gdiv((long long)N_NODES * 32, 256), 256, 0, stream>>>(
      x, actA, N_NODES, 32, N_NODES, 32);
  {
    int pi = 4;  // node_layers[0]
    run_gat_layer(stream, actA, 32, P(pi), P(pi + 1), P(pi + 2), P(pi + 3), Wbuf,
                  hW, sS, sD, mEnc, den, earr, agg, src, dstp, hf, actA);
    for (int l = 1; l < 4; ++l) {
      pi = 4 + 4 * l;
      run_gat_layer(stream, actA, 128, P(pi), P(pi + 1), P(pi + 2), P(pi + 3), Wbuf,
                    hW, sS, sD, mEnc, den, earr, agg, src, dstp, hf,
                    (l < 3) ? actA : (_Float16*)nullptr);
    }
  }

  // --- h = concat(x, f) -> gat_layers chain ---
  concat_xf_kernel<<<gdiv((long long)N_NODES * 160, 256), 256, 0, stream>>>(x, hf, actA,
                                                                            N_NODES, N_NODES);
  {
    int pi = 20;  // gat_layers[0], fin = 160
    run_gat_layer(stream, actA, 160, P(pi), P(pi + 1), P(pi + 2), P(pi + 3), Wbuf,
                  hW, sS, sD, mEnc, den, earr, agg, src, dstp, hf, actA);
    for (int l = 1; l < 4; ++l) {
      pi = 20 + 4 * l;
      run_gat_layer(stream, actA, 128, P(pi), P(pi + 1), P(pi + 2), P(pi + 3), Wbuf,
                    hW, sS, sD, mEnc, den, earr, agg, src, dstp, hf, actA);
    }
  }
  // now: hf = h (fp32, N x 128), actA = h (f16, N x 128)

  // --- global mean pool -> pi_fc MLP ---
  fill_u32_kernel<<<gdiv(BGR * 128, 256), 256, 0, stream>>>((unsigned*)pool, 0u, BGR * 128);
  fill_u32_kernel<<<gdiv(BGR, 256), 256, 0, stream>>>((unsigned*)counts, 0u, BGR);
  count_kernel<<<gdiv(N_NODES, 256), 256, 0, stream>>>(batch, counts, N_NODES);
  pool_accum_kernel<<<gdiv(N_NODES, 8), 256, 0, stream>>>(batch, hf, pool, N_NODES);
  pool_fin_kernel<<<gdiv(BGR * 128, 256), 256, 0, stream>>>(pool, counts, bufB1);

  run_gemm(stream, Wbuf, bufB1, P(46), P(47), 128, 128, 512, 512, BGR, BGR, 1,
           nullptr, 0, bufB2, 512);
  run_gemm(stream, Wbuf, bufB2, P(48), P(49), 512, 512, 512, 512, BGR, BGR, 1,
           nullptr, 0, bufB1, 512);
  run_gemm(stream, Wbuf, bufB1, P(50), P(51), 512, 512, 512, 512, BGR, BGR, 1,
           nullptr, 0, bufB2, 512);
  run_gemm(stream, Wbuf, bufB2, P(52), P(53), 512, 512, 256, 256, BGR, BGR, 1,
           nullptr, 0, bufB1, 256);
  run_gemm(stream, Wbuf, bufB1, P(54), P(55), 256, 256, 10, 16, BGR, BGR, 0,
           pi_out, 10, nullptr, 0);

  // --- metal embedding (relu after both linears) + query ---
  convert_pad_kernel<<<gdiv(BGR * 32, 256), 256, 0, stream>>>(metal, bufB1, BGR, 8, BGR, 32);
  run_gemm(stream, Wbuf, bufB1, P(36), P(37), 8, 32, 128, 128, BGR, BGR, 1,
           nullptr, 0, bufB2, 128);
  run_gemm(stream, Wbuf, bufB2, P(38), P(39), 128, 128, 128, 128, BGR, BGR, 1,
           membed, 128, bufB1, 128);
  run_gemm(stream, Wbuf, bufB1, P(40), P(41), 128, 128, 128, 128, BGR, BGR, 0,
           qbuf, 128, nullptr, 0);

  // --- k, v projections (reuse hW / agg buffers) ---
  run_gemm(stream, Wbuf, actA, P(42), P(43), 128, 128, 128, 128, N_NODES, N_NODES, 0,
           hW, 128, nullptr, 0);
  run_gemm(stream, Wbuf, actA, P(44), P(45), 128, 128, 128, 128, N_NODES, N_NODES, 0,
           agg, 128, nullptr, 0);

  // --- batched cross attention ---
  fill_u32_kernel<<<gdiv(BGR, 256), 256, 0, stream>>>(mB, 0u, BGR);
  fill_u32_kernel<<<gdiv(BGR, 256), 256, 0, stream>>>((unsigned*)denB, 0u, BGR);
  fill_u32_kernel<<<gdiv(BGR * 128, 256), 256, 0, stream>>>((unsigned*)attnB, 0u, BGR * 128);
  attn_score_kernel<<<gdiv(N_NODES, 8), 256, 0, stream>>>(qbuf, hW, batch, sS, mB, N_NODES);
  attn_exp_kernel<<<gdiv(N_NODES, 256), 256, 0, stream>>>(batch, sS, mB, denB, N_NODES);
  attn_agg_kernel<<<gdiv(N_NODES, 8), 256, 0, stream>>>(batch, sS, denB, agg, attnB, N_NODES);

  // --- combined = [pi_out | attn | metal_embed], regressor MLP ---
  combine_kernel<<<gdiv(BGR * 288, 256), 256, 0, stream>>>(pi_out, attnB, membed, bufB1);
  run_gemm(stream, Wbuf, bufB1, P(56), P(57), 266, 288, 512, 512, BGR, BGR, 1,
           nullptr, 0, bufB2, 512);
  run_gemm(stream, Wbuf, bufB2, P(58), P(59), 512, 512, 512, 512, BGR, BGR, 1,
           nullptr, 0, bufB1, 512);
  run_gemm(stream, Wbuf, bufB1, P(60), P(61), 512, 512, 512, 512, BGR, BGR, 1,
           nullptr, 0, bufB2, 512);
  run_gemm(stream, Wbuf, bufB2, P(62), P(63), 512, 512, 256, 256, BGR, BGR, 1,
           nullptr, 0, bufB1, 256);
  run_gemm(stream, Wbuf, bufB1, P(64), P(65), 256, 256, 1, 16, BGR, BGR, 0,
           energy, 1, nullptr, 0);

  final_kernel<<<gdiv(BGR + BGR * 10, 256), 256, 0, stream>>>(energy, pi_out, (float*)d_out);
}
